// DummyTransformer_46239617909186
// MI455X (gfx1250) — 8-bit matrix operand order repaired and checked by reading, compiled, not run on hardware
//
#include <hip/hip_runtime.h>
#include <cstdint>


typedef int v8i __attribute__((ext_vector_type(8)));
typedef int v4i __attribute__((ext_vector_type(4)));

#define AS1 __attribute__((address_space(1)))
#define AS3 __attribute__((address_space(3)))

#if defined(__AMDGCN__) && __has_builtin(__builtin_amdgcn_global_load_async_to_lds_b128)
#define USE_ASYNC_LDS 1
#else
#define USE_ASYNC_LDS 0
#endif

#if USE_ASYNC_LDS
static __device__ __forceinline__ AS1 v4i* as_global(const void* p) {
  return (AS1 v4i*)(AS1 void*)const_cast<void*>(p);
}
static __device__ __forceinline__ AS3 v4i* as_lds(void* p) {
  return (AS3 v4i*)(AS3 void*)p;
}
static __device__ __forceinline__ void wait_async0() {
#if __has_builtin(__builtin_amdgcn_s_wait_asynccnt)
  __builtin_amdgcn_s_wait_asynccnt(0);
#else
  asm volatile("s_wait_asynccnt 0" ::: "memory");
#endif
}
#endif

#define CB   4
#define CS   1024
#define CE   1024
#define CM   4096
#define CH   16
#define CHD  64
#define NTOK 4096

enum {
  R_X = 0, R_WQ, R_WK, R_WV, R_WO, R_W1, R_W2,
  R_BQ, R_BK, R_BV, R_BO, R_B1, R_B2,
  R_Q, R_K, R_V, R_ATT, R_AO, R_X1, R_H1, R_H2,
  E_X = 32, E_W, E_Q, E_K, E_V, E_ATT, E_AO, E_X1, E_H, E_H2
};

__global__ void k_zero_sc(float* p) {
  int t = threadIdx.x;
  if (t < 64) p[t] = 0.f;
}

template <bool RELU>
__global__ __launch_bounds__(256) void k_absmax(const float* __restrict__ x,
                                                long n, float* __restrict__ slot) {
  float m = 0.f;
  for (long i = (long)blockIdx.x * blockDim.x + threadIdx.x; i < n;
       i += (long)gridDim.x * blockDim.x) {
    float v = x[i];
    v = RELU ? fmaxf(v, 0.f) : fabsf(v);
    m = fmaxf(m, v);
  }
#pragma unroll
  for (int o = 16; o > 0; o >>= 1) m = fmaxf(m, __shfl_xor(m, o, 32));
  __shared__ float sm[8];
  if ((threadIdx.x & 31) == 0) sm[threadIdx.x >> 5] = m;
  __syncthreads();
  if (threadIdx.x < 8) {
    m = sm[threadIdx.x];
#pragma unroll
    for (int o = 4; o > 0; o >>= 1) m = fmaxf(m, __shfl_xor(m, o, 32));
    if (threadIdx.x == 0) atomicMax((int*)slot, __float_as_int(m));
  }
}

template <int MODE>
__global__ __launch_bounds__(256) void k_quant(const float* __restrict__ src,
                                               int8_t* __restrict__ dst,
                                               float* __restrict__ SC, int raw,
                                               int eff, float pre, long n) {
  const float qmax = (MODE == 2) ? 255.f : 127.f;
  const float qmin = (MODE == 0) ? -128.f : ((MODE == 1) ? -127.f : 0.f);
  const float s = fmaxf(SC[raw] * pre / qmax, 1e-8f);
  if (blockIdx.x == 0 && threadIdx.x == 0) SC[eff] = s;
  const float inv = 1.f / s;
  for (long i = (long)blockIdx.x * blockDim.x + threadIdx.x; i < n;
       i += (long)gridDim.x * blockDim.x) {
    float q = fminf(fmaxf(rintf(src[i] * pre * inv), qmin), qmax);
    dst[i] = (int8_t)(int)q;
  }
}

template <int VT>
__global__ __launch_bounds__(256) void k_quant_head(const float* __restrict__ src,
                                                    int8_t* __restrict__ dst,
                                                    float* __restrict__ SC,
                                                    int raw, int eff, float pre) {
  const float s = fmaxf(SC[raw] * pre * (1.f / 127.f), 1e-8f);
  if (blockIdx.x == 0 && threadIdx.x == 0) SC[eff] = s;
  const float inv = 1.f / s;
  const long n = (long)NTOK * CE;
  for (long i = (long)blockIdx.x * blockDim.x + threadIdx.x; i < n;
       i += (long)gridDim.x * blockDim.x) {
    int e = (int)(i & (CE - 1));
    long tok = i >> 10;
    int h = e >> 6, d = e & 63;
    int b = (int)(tok >> 10), sp = (int)(tok & (CS - 1));
    float q = fminf(fmaxf(rintf(src[i] * pre * inv), -128.f), 127.f);
    long di = VT ? ((((long)(b * CH + h)) * CHD + d) * CS + sp)
                 : ((((long)(b * CH + h)) * CS + sp) * CHD + d);
    dst[di] = (int8_t)(int)q;
  }
}

__global__ __launch_bounds__(256) void k_posadd(const float* __restrict__ x,
                                                const float* __restrict__ SC,
                                                float* __restrict__ out) {
  const float sx = fmaxf(SC[R_X] * (1.f / 127.f), 1e-8f);
  const float invx = 1.f / sx;
  const long n = (long)NTOK * CE;
  for (long i = (long)blockIdx.x * blockDim.x + threadIdx.x; i < n;
       i += (long)gridDim.x * blockDim.x) {
    int e = (int)(i & (CE - 1));
    int sp = (int)((i >> 10) & (CS - 1));
    float xq = fminf(fmaxf(rintf(x[i] * invx), -128.f), 127.f) * sx;
    float f = __powf(10000.f, -((float)(e & ~1) / (float)CE));
    float ang = f * (float)sp;
    float p = (e & 1) ? __cosf(ang) : __sinf(ang);
    float pq = fminf(fmaxf(rintf(p * 127.f), -128.f), 127.f) * (1.f / 127.f);
    out[i] = xq + pq;
  }
}

__global__ __launch_bounds__(256) void k_add_ln(const int8_t* __restrict__ a8,
                                                const int8_t* __restrict__ b8,
                                                const float* __restrict__ SC,
                                                int sa, int sb,
                                                float* __restrict__ out) {
  const int t = threadIdx.x;
  const float s1 = SC[sa], s2 = SC[sb];
  const size_t base = (size_t)blockIdx.x * CE;
  float v[4];
  float sum = 0.f;
#pragma unroll
  for (int j = 0; j < 4; j++) {
    int e = t + j * 256;
    float xv = (float)a8[base + e] * s1 + (float)b8[base + e] * s2;
    v[j] = xv;
    sum += xv;
  }
  __shared__ float red[9];
#pragma unroll
  for (int o = 16; o > 0; o >>= 1) sum += __shfl_xor(sum, o, 32);
  if ((t & 31) == 0) red[t >> 5] = sum;
  __syncthreads();
  if (t == 0) {
    float tot = 0.f;
    for (int k = 0; k < 8; k++) tot += red[k];
    red[8] = tot * (1.f / CE);
  }
  __syncthreads();
  const float mu = red[8];
  __syncthreads();
  float var = 0.f;
#pragma unroll
  for (int j = 0; j < 4; j++) {
    float d = v[j] - mu;
    var += d * d;
  }
#pragma unroll
  for (int o = 16; o > 0; o >>= 1) var += __shfl_xor(var, o, 32);
  if ((t & 31) == 0) red[t >> 5] = var;
  __syncthreads();
  if (t == 0) {
    float tot = 0.f;
    for (int k = 0; k < 8; k++) tot += red[k];
    red[8] = tot * (1.f / CE);
  }
  __syncthreads();
  const float rs = rsqrtf(red[8] + 1e-5f);
#pragma unroll
  for (int j = 0; j < 4; j++) out[base + t + j * 256] = (v[j] - mu) * rs;
}

template <bool SGN_A>
__global__ __launch_bounds__(256) void k_gemm(const int8_t* __restrict__ A,
                                              const int8_t* __restrict__ W,
                                              const float* __restrict__ SC,
                                              int sA_i, int sW_i, int bMax_i,
                                              const float* __restrict__ bias,
                                              float* __restrict__ C, int K, int N) {
  __shared__ __align__(16) int8_t lA[2][128 * 64];
  __shared__ __align__(16) int8_t lB[2][128 * 64];
  const int tid = threadIdx.x;
  const int wave = tid >> 5, lane = tid & 31;
  const int lm = lane & 15, lh = lane >> 4;
  const size_t m0 = (size_t)blockIdx.x * 128;
  const size_t n0 = (size_t)blockIdx.y * 128;
  v8i zero = {0, 0, 0, 0, 0, 0, 0, 0};
  v8i acc[8];
#pragma unroll
  for (int i = 0; i < 8; i++) acc[i] = zero;
  const int c0 = tid * 2;
  const int r0 = c0 >> 2, cb0 = (c0 & 3) << 4;
  const int r1 = (c0 + 1) >> 2, cb1 = ((c0 + 1) & 3) << 4;

  auto stage = [&](int k0, int bsel) {
#if USE_ASYNC_LDS
    __builtin_amdgcn_global_load_async_to_lds_b128(
        as_global(A + (m0 + r0) * K + k0 + cb0), as_lds(&lA[bsel][r0 * 64 + cb0]), 0, 0);
    __builtin_amdgcn_global_load_async_to_lds_b128(
        as_global(A + (m0 + r1) * K + k0 + cb1), as_lds(&lA[bsel][r1 * 64 + cb1]), 0, 0);
    __builtin_amdgcn_global_load_async_to_lds_b128(
        as_global(W + (n0 + r0) * K + k0 + cb0), as_lds(&lB[bsel][r0 * 64 + cb0]), 0, 0);
    __builtin_amdgcn_global_load_async_to_lds_b128(
        as_global(W + (n0 + r1) * K + k0 + cb1), as_lds(&lB[bsel][r1 * 64 + cb1]), 0, 0);
#else
    *(int4*)&lA[bsel][r0 * 64 + cb0] = *(const int4*)&A[(m0 + r0) * K + k0 + cb0];
    *(int4*)&lA[bsel][r1 * 64 + cb1] = *(const int4*)&A[(m0 + r1) * K + k0 + cb1];
    *(int4*)&lB[bsel][r0 * 64 + cb0] = *(const int4*)&W[(n0 + r0) * K + k0 + cb0];
    *(int4*)&lB[bsel][r1 * 64 + cb1] = *(const int4*)&W[(n0 + r1) * K + k0 + cb1];
#endif
  };

  stage(0, 0);
#if USE_ASYNC_LDS
  wait_async0();
#endif
  __syncthreads();

  const int nk = K >> 6;
  int buf = 0;
  for (int kk = 0; kk < nk; ++kk) {
    if (kk + 1 < nk) stage((kk + 1) << 6, buf ^ 1);
    v8i af;
#pragma unroll
    for (int j = 0; j < 8; j++) {
      int kb = ((j & 4) << 3) + (lh << 4) + ((j & 3) << 2);
      af[j] = *(const int*)&lA[buf][(wave * 16 + lm) * 64 + kb];
    }
#pragma unroll
    for (int nt = 0; nt < 8; ++nt) {
      v8i bf;
#pragma unroll
      for (int j = 0; j < 8; j++) {
        int kb = ((j & 4) << 3) + (lh << 4) + ((j & 3) << 2);
        bf[j] = *(const int*)&lB[buf][(nt * 16 + lm) * 64 + kb];
      }
      acc[nt] = __builtin_amdgcn_wmma_i32_16x16x64_iu8(SGN_A, af, true, bf,
                                                       acc[nt], false, false);
    }
#if USE_ASYNC_LDS
    wait_async0();
#endif
    __syncthreads();
    buf ^= 1;
  }

  const float sAW = SC[sA_i] * SC[sW_i];
  const float sb = fmaxf(SC[bMax_i] * (1.f / 127.f), 1e-8f);
#pragma unroll
  for (int nt = 0; nt < 8; ++nt) {
    int n = (int)n0 + nt * 16 + lm;
    float bq = fminf(fmaxf(rintf(bias[n] / sb), -128.f), 127.f) * sb;
#pragma unroll
    for (int r = 0; r < 8; r++) {
      size_t mrow = m0 + wave * 16 + lh * 8 + r;
      C[mrow * (size_t)N + n] = (float)acc[nt][r] * sAW + bq;
    }
  }
}

__global__ __launch_bounds__(256) void k_attn(const int8_t* __restrict__ q8,
                                              const int8_t* __restrict__ k8,
                                              const int8_t* __restrict__ v8t,
                                              const float* __restrict__ SC,
                                              int sq_i, int sk_i, int sv_i,
                                              float* __restrict__ out) {
  __shared__ __align__(16) int8_t lw[8][16 * 64];
  const int bh = blockIdx.x >> 3;
  const int qt = blockIdx.x & 7;
  const int wave = threadIdx.x >> 5, lane = threadIdx.x & 31;
  const int lm = lane & 15, lh = lane >> 4;
  const int qb = qt * 128 + wave * 16;
  const int b = bh >> 4, h = bh & 15;
  const int8_t* qp = q8 + (size_t)bh * (CS * CHD);
  const int8_t* kp = k8 + (size_t)bh * (CS * CHD);
  const int8_t* vp = v8t + (size_t)bh * (CHD * CS);
  const float sQK = SC[sq_i] * SC[sk_i];
  v8i zero = {0, 0, 0, 0, 0, 0, 0, 0};
  v8i qa;
#pragma unroll
  for (int j = 0; j < 8; j++) {
    int kb = ((j & 4) << 3) + (lh << 4) + ((j & 3) << 2);
    qa[j] = *(const int*)&qp[(qb + lm) * CHD + kb];
  }
  float m[8], l[8];
#pragma unroll
  for (int r = 0; r < 8; r++) { m[r] = -1e30f; l[r] = 0.f; }
  for (int kt = 0; kt < 64; ++kt) {
    __builtin_prefetch(&kp[((kt + 1) & 63) * (16 * CHD) + lane * 32], 0, 0);
    v8i kf;
#pragma unroll
    for (int j = 0; j < 8; j++) {
      int kb = ((j & 4) << 3) + (lh << 4) + ((j & 3) << 2);
      kf[j] = *(const int*)&kp[(kt * 16 + lm) * CHD + kb];
    }
    v8i c = __builtin_amdgcn_wmma_i32_16x16x64_iu8(true, qa, true, kf, zero,
                                                   false, false);
#pragma unroll
    for (int r = 0; r < 8; r++) {
      int qrow = qb + lh * 8 + r;
      int kcol = kt * 16 + lm;
      float s = (kcol <= qrow) ? (float)c[r] * sQK : -1e30f;
      float mn = fmaxf(m[r], s);
      l[r] = l[r] * __expf(m[r] - mn) + __expf(s - mn);
      m[r] = mn;
    }
  }
#pragma unroll
  for (int r = 0; r < 8; r++) {
#pragma unroll
    for (int msk = 1; msk < 16; msk <<= 1) {
      float mo = __shfl_xor(m[r], msk, 32);
      float lo = __shfl_xor(l[r], msk, 32);
      float mn = fmaxf(m[r], mo);
      l[r] = l[r] * __expf(m[r] - mn) + lo * __expf(mo - mn);
      m[r] = mn;
    }
  }
  float invl[8];
#pragma unroll
  for (int r = 0; r < 8; r++) invl[r] = 1.f / l[r];
  v8i vacc[4];
#pragma unroll
  for (int t = 0; t < 4; t++) vacc[t] = zero;
  for (int kg = 0; kg < 16; ++kg) {
#pragma unroll
    for (int j4 = 0; j4 < 4; ++j4) {
      int kt = kg * 4 + j4;
      v8i kf;
#pragma unroll
      for (int j = 0; j < 8; j++) {
        int kb = ((j & 4) << 3) + (lh << 4) + ((j & 3) << 2);
        kf[j] = *(const int*)&kp[(kt * 16 + lm) * CHD + kb];
      }
      v8i c = __builtin_amdgcn_wmma_i32_16x16x64_iu8(true, qa, true, kf, zero,
                                                     false, false);
#pragma unroll
      for (int r = 0; r < 8; r++) {
        int qrow = qb + lh * 8 + r;
        int kcol = kt * 16 + lm;
        float s = (kcol <= qrow) ? (float)c[r] * sQK : -1e30f;
        float w = __expf(s - m[r]) * invl[r];
        int iw = (int)fminf(fmaxf(rintf(w * 127.f), 0.f), 127.f);
        lw[wave][(lh * 8 + r) * 64 + j4 * 16 + lm] = (int8_t)iw;
      }
    }
    __syncthreads();
    v8i wa;
#pragma unroll
    for (int j = 0; j < 8; j++) {
      int kb = ((j & 4) << 3) + (lh << 4) + ((j & 3) << 2);
      wa[j] = *(const int*)&lw[wave][lm * 64 + kb];
    }
#pragma unroll
    for (int nt = 0; nt < 4; ++nt) {
      v8i vf;
#pragma unroll
      for (int j = 0; j < 8; j++) {
        int kb = ((j & 4) << 3) + (lh << 4) + ((j & 3) << 2);
        vf[j] = *(const int*)&vp[(size_t)(nt * 16 + lm) * CS + kg * 64 + kb];
      }
      vacc[nt] = __builtin_amdgcn_wmma_i32_16x16x64_iu8(true, wa, true, vf,
                                                        vacc[nt], false, false);
    }
    __syncthreads();
  }
  const float sOut = SC[sv_i] * (1.f / 127.f);
#pragma unroll
  for (int nt = 0; nt < 4; ++nt) {
#pragma unroll
    for (int r = 0; r < 8; r++) {
      int qrow = qb + lh * 8 + r;
      int d = nt * 16 + lm;
      out[((size_t)(b * CS + qrow)) * CE + h * CHD + d] = (float)vacc[nt][r] * sOut;
    }
  }
}

static constexpr size_t FB = (size_t)NTOK * CE * 4;
static constexpr size_t I8E = (size_t)NTOK * CE;
static constexpr size_t OFF_SC = 0;
static constexpr size_t OFF_XF = 4096;
static constexpr size_t OFF_QF = OFF_XF + FB;
static constexpr size_t OFF_KF = OFF_QF + FB;
static constexpr size_t OFF_VF = OFF_KF + FB;
static constexpr size_t OFF_AF = OFF_VF + FB;
static constexpr size_t OFF_H1 = OFF_AF + FB;
static constexpr size_t OFF_I8X = OFF_H1 + (size_t)NTOK * CM * 4;
static constexpr size_t OFF_I8B = OFF_I8X + I8E;
static constexpr size_t OFF_I8H = OFF_I8B + I8E;
static constexpr size_t OFF_I8Q = OFF_I8H + (size_t)NTOK * CM;
static constexpr size_t OFF_I8K = OFF_I8Q + I8E;
static constexpr size_t OFF_I8V = OFF_I8K + I8E;
static constexpr size_t OFF_W8 = OFF_I8V + I8E;

static inline int gsz(long n) {
  long g = (n + 255) / 256;
  return (int)(g > 4096 ? 4096 : g);
}

extern "C" void kernel_launch(void* const* d_in, const int* in_sizes, int n_in,
                              void* d_out, int out_size, void* d_ws, size_t ws_size,
                              hipStream_t stream) {
  (void)in_sizes; (void)n_in; (void)out_size; (void)ws_size;
  const float* x  = (const float*)d_in[0];
  const float* Wq = (const float*)d_in[1];
  const float* Wk = (const float*)d_in[2];
  const float* Wv = (const float*)d_in[3];
  const float* Wo = (const float*)d_in[4];
  const float* bq = (const float*)d_in[5];
  const float* bk = (const float*)d_in[6];
  const float* bv = (const float*)d_in[7];
  const float* bo = (const float*)d_in[8];
  const float* W1 = (const float*)d_in[9];
  const float* b1 = (const float*)d_in[10];
  const float* W2 = (const float*)d_in[11];
  const float* b2 = (const float*)d_in[12];

  char* w = (char*)d_ws;
  float* SC  = (float*)(w + OFF_SC);
  float* XF  = (float*)(w + OFF_XF);
  float* QF  = (float*)(w + OFF_QF);
  float* KF  = (float*)(w + OFF_KF);
  float* VF  = (float*)(w + OFF_VF);
  float* AF  = (float*)(w + OFF_AF);
  float* H1F = (float*)(w + OFF_H1);
  int8_t* I8X = (int8_t*)(w + OFF_I8X);
  int8_t* I8B = (int8_t*)(w + OFF_I8B);
  int8_t* I8H = (int8_t*)(w + OFF_I8H);
  int8_t* I8Q = (int8_t*)(w + OFF_I8Q);
  int8_t* I8K = (int8_t*)(w + OFF_I8K);
  int8_t* I8V = (int8_t*)(w + OFF_I8V);
  int8_t* W8  = (int8_t*)(w + OFF_W8);

  const long nXE = (long)NTOK * CE;
  const long nXM = (long)NTOK * CM;
  const dim3 gN1024(NTOK / 128, CE / 128);
  const dim3 gN4096(NTOK / 128, CM / 128);

  k_zero_sc<<<1, 64, 0, stream>>>(SC);
  k_absmax<false><<<gsz(nXE), 256, 0, stream>>>(x, nXE, SC + R_X);
  k_posadd<<<gsz(nXE), 256, 0, stream>>>(x, SC, XF);

  for (int l = 0; l < 4; ++l) {
    const float* Wq_l = Wq + (size_t)l * CE * CE;
    const float* Wk_l = Wk + (size_t)l * CE * CE;
    const float* Wv_l = Wv + (size_t)l * CE * CE;
    const float* Wo_l = Wo + (size_t)l * CE * CE;
    const float* W1_l = W1 + (size_t)l * CM * CE;
    const float* W2_l = W2 + (size_t)l * CE * CM;
    const float* bq_l = bq + (size_t)l * CE;
    const float* bk_l = bk + (size_t)l * CE;
    const float* bv_l = bv + (size_t)l * CE;
    const float* bo_l = bo + (size_t)l * CE;
    const float* b1_l = b1 + (size_t)l * CM;
    const float* b2_l = b2 + (size_t)l * CE;

    k_zero_sc<<<1, 64, 0, stream>>>(SC);
    k_absmax<false><<<gsz(nXE), 256, 0, stream>>>(XF, nXE, SC + R_X);
    k_absmax<false><<<gsz(CE * CE), 256, 0, stream>>>(Wq_l, (long)CE * CE, SC + R_WQ);
    k_absmax<false><<<gsz(CE * CE), 256, 0, stream>>>(Wk_l, (long)CE * CE, SC + R_WK);
    k_absmax<false><<<gsz(CE * CE), 256, 0, stream>>>(Wv_l, (long)CE * CE, SC + R_WV);
    k_absmax<false><<<gsz(CE * CE), 256, 0, stream>>>(Wo_l, (long)CE * CE, SC + R_WO);
    k_absmax<false><<<gsz((long)CM * CE), 256, 0, stream>>>(W1_l, (long)CM * CE, SC + R_W1);
    k_absmax<false><<<gsz((long)CE * CM), 256, 0, stream>>>(W2_l, (long)CE * CM, SC + R_W2);
    k_absmax<false><<<gsz(CE), 256, 0, stream>>>(bq_l, CE, SC + R_BQ);
    k_absmax<false><<<gsz(CE), 256, 0, stream>>>(bk_l, CE, SC + R_BK);
    k_absmax<false><<<gsz(CE), 256, 0, stream>>>(bv_l, CE, SC + R_BV);
    k_absmax<false><<<gsz(CE), 256, 0, stream>>>(bo_l, CE, SC + R_BO);
    k_absmax<false><<<gsz(CM), 256, 0, stream>>>(b1_l, CM, SC + R_B1);
    k_absmax<false><<<gsz(CE), 256, 0, stream>>>(b2_l, CE, SC + R_B2);

    k_quant<0><<<gsz(nXE), 256, 0, stream>>>(XF, I8X, SC, R_X, E_X, 1.f, nXE);

    k_quant<1><<<gsz(CE * CE), 256, 0, stream>>>(Wq_l, W8, SC, R_WQ, E_W, 1.f, (long)CE * CE);
    k_gemm<true><<<gN1024, 256, 0, stream>>>(I8X, W8, SC, E_X, E_W, R_BQ, bq_l, QF, CE, CE);
    k_quant<1><<<gsz(CE * CE), 256, 0, stream>>>(Wk_l, W8, SC, R_WK, E_W, 1.f, (long)CE * CE);
    k_gemm<true><<<gN1024, 256, 0, stream>>>(I8X, W8, SC, E_X, E_W, R_BK, bk_l, KF, CE, CE);
    k_quant<1><<<gsz(CE * CE), 256, 0, stream>>>(Wv_l, W8, SC, R_WV, E_W, 1.f, (long)CE * CE);
    k_gemm<true><<<gN1024, 256, 0, stream>>>(I8X, W8, SC, E_X, E_W, R_BV, bv_l, VF, CE, CE);

    k_absmax<false><<<gsz(nXE), 256, 0, stream>>>(QF, nXE, SC + R_Q);
    k_absmax<false><<<gsz(nXE), 256, 0, stream>>>(KF, nXE, SC + R_K);
    k_absmax<false><<<gsz(nXE), 256, 0, stream>>>(VF, nXE, SC + R_V);
    k_quant_head<0><<<gsz(nXE), 256, 0, stream>>>(QF, I8Q, SC, R_Q, E_Q, 0.125f);
    k_quant_head<0><<<gsz(nXE), 256, 0, stream>>>(KF, I8K, SC, R_K, E_K, 1.f);
    k_quant_head<1><<<gsz(nXE), 256, 0, stream>>>(VF, I8V, SC, R_V, E_V, 1.f);

    k_attn<<<CB * CH * (CS / 128), 256, 0, stream>>>(I8Q, I8K, I8V, SC, E_Q, E_K, E_V, AF);

    k_absmax<false><<<gsz(nXE), 256, 0, stream>>>(AF, nXE, SC + R_ATT);
    k_quant<0><<<gsz(nXE), 256, 0, stream>>>(AF, I8B, SC, R_ATT, E_ATT, 1.f, nXE);
    k_quant<1><<<gsz(CE * CE), 256, 0, stream>>>(Wo_l, W8, SC, R_WO, E_W, 1.f, (long)CE * CE);
    k_gemm<true><<<gN1024, 256, 0, stream>>>(I8B, W8, SC, E_ATT, E_W, R_BO, bo_l, QF, CE, CE);

    k_absmax<false><<<gsz(nXE), 256, 0, stream>>>(QF, nXE, SC + R_AO);
    k_quant<0><<<gsz(nXE), 256, 0, stream>>>(QF, I8B, SC, R_AO, E_AO, 1.f, nXE);
    k_add_ln<<<NTOK, 256, 0, stream>>>(I8X, I8B, SC, E_X, E_AO, XF);

    k_absmax<false><<<gsz(nXE), 256, 0, stream>>>(XF, nXE, SC + R_X1);
    k_quant<0><<<gsz(nXE), 256, 0, stream>>>(XF, I8X, SC, R_X1, E_X1, 1.f, nXE);
    k_quant<1><<<gsz((long)CM * CE), 256, 0, stream>>>(W1_l, W8, SC, R_W1, E_W, 1.f, (long)CM * CE);
    k_gemm<true><<<gN4096, 256, 0, stream>>>(I8X, W8, SC, E_X1, E_W, R_B1, b1_l, H1F, CE, CM);
    k_absmax<true><<<gsz(nXM), 256, 0, stream>>>(H1F, nXM, SC + R_H1);
    k_quant<2><<<gsz(nXM), 256, 0, stream>>>(H1F, I8H, SC, R_H1, E_H, 1.f, nXM);
    k_quant<1><<<gsz((long)CE * CM), 256, 0, stream>>>(W2_l, W8, SC, R_W2, E_W, 1.f, (long)CE * CM);
    k_gemm<false><<<gN1024, 256, 0, stream>>>(I8H, W8, SC, E_H, E_W, R_B2, b2_l, QF, CM, CE);

    k_absmax<false><<<gsz(nXE), 256, 0, stream>>>(QF, nXE, SC + R_H2);
    k_quant<0><<<gsz(nXE), 256, 0, stream>>>(QF, I8B, SC, R_H2, E_H2, 1.f, nXE);
    float* dst = (l == 3) ? (float*)d_out : XF;
    k_add_ln<<<NTOK, 256, 0, stream>>>(I8X, I8B, SC, E_X1, E_H2, dst);
  }
}
